// CalibreLoss_12781822672953
// MI455X (gfx1250) — matrix operand order repaired and checked by reading, not compiled, not run on hardware
//
#include <hip/hip_runtime.h>
#include <hip/hip_bf16.h>

// ---------------------------------------------------------------------------
// CalibreLoss on MI455X (gfx1250, wave32).
// Dominant cost: feat @ feat.T (16384 x 16384 x 128) = 68.7 GFLOP, fused
// (never materialized) via v_wmma_f32_16x16x32_f16 with streaming softmax
// accumulation. B tiles are software-pipelined (double-buffered in VGPRs) so
// global_load_b128 latency overlaps WMMA + epilogue work; launch bounds lift
// the VGPR cap so A/B tiles stay resident (no scratch spills). Small kernels
// stage their 32KB LDS tables with gfx1250 async global->LDS copies.
// ---------------------------------------------------------------------------

typedef __attribute__((ext_vector_type(16))) _Float16 v16h;
typedef __attribute__((ext_vector_type(8)))  _Float16 v8h;
typedef __attribute__((ext_vector_type(8)))  float    v8f;

#define BB   8192
#define DD   128
#define KK   64
#define NN   16384           // 2*B
#define INV_T 14.2857142857f // 1/0.07
#define C_EXP 20.6096707744f // INV_T * log2(e)

// ---- workspace layout (bytes) ----
#define OFF_FEAT   0u          // 16384*128 f16  = 4 MiB
#define OFF_ASSIGN 4194304u    // 16384 int
#define OFF_SUMA   4259840u    // K*D f32
#define OFF_SUMB   4292608u
#define OFF_SUMS   4325376u
#define OFF_CNTA   4358144u    // 64 f32
#define OFF_CNTB   4358400u
#define OFF_HIST   4358656u    // 16 f32 (10 used)
#define OFF_DENOM  4358720u    // 16384 f32
#define OFF_POS    4424256u    // 16384 f32
#define OFF_ACC    4489792u    // acc[0]=lp_sum acc[1]=ln_sum
#define OFF_ZBUF   4489856u    // 128*128 f32 (normalized protos)
#define OFF_SUP    4555392u    // 64*128 f32
#define ZERO_FLOATS 57504u     // (4489856-4259840)/4

__device__ __forceinline__ float wave_red_add(float v, int lastMask) {
    for (int m = 1; m <= lastMask; m <<= 1) v += __shfl_xor(v, m, 32);
    return v;
}

// Async-stage a 64x128 f32 table into LDS, transposed to lds[d*64+k].
// The destination LDS array is the kernel's only __shared__ object -> LDS
// offset 0, so the per-lane LDS address is just the element byte offset.
// Tracked by ASYNCcnt; caller must __syncthreads() after.
__device__ __forceinline__ void async_stage_transposed(const float* __restrict__ src,
                                                       int tid) {
    for (int t = tid; t < KK * DD; t += 256) {
        int k = t >> 7, d = t & 127;
        unsigned ldsoff = (unsigned)((d * KK + k) * 4);
        unsigned long long ga = (unsigned long long)(const void*)(src + t);
        asm volatile("global_load_async_to_lds_b32 %0, %1, off"
                     :: "v"(ldsoff), "v"(ga) : "memory");
    }
    asm volatile("s_wait_asynccnt 0x0" ::: "memory");
}

// ---- 0: zero accumulator region ----
__global__ void k_zero(float* p, unsigned n) {
    unsigned t = blockIdx.x * blockDim.x + threadIdx.x;
    if (t < n) p[t] = 0.0f;
}

// ---- 1: cluster assignment: argmin_k || x - c_k ||^2 (wave per row) ----
__global__ void k_assign(const float* __restrict__ ea, const float* __restrict__ eb,
                         const float* __restrict__ centers, int* __restrict__ assign) {
    __shared__ float cs[KK * DD]; // transposed: cs[d*64 + k]
    int tid = threadIdx.x;
    async_stage_transposed(centers, tid);
    __syncthreads();
    int wave = tid >> 5, lane = tid & 31;
    int row = blockIdx.x * 8 + wave;
    const float* x = (row < BB) ? (ea + (size_t)row * DD) : (eb + (size_t)(row - BB) * DD);
    int k0 = lane, k1 = lane + 32;
    float a0 = 0.0f, a1 = 0.0f;
    for (int d = 0; d < DD; ++d) {
        float xd = x[d];
        float f0 = xd - cs[d * KK + k0];
        float f1 = xd - cs[d * KK + k1];
        a0 += f0 * f0; a1 += f1 * f1;
    }
    float bv = a0; int bk = k0;
    if (a1 < bv) { bv = a1; bk = k1; }
    for (int m = 16; m >= 1; m >>= 1) {
        float ov = __shfl_xor(bv, m, 32);
        int   ok = __shfl_xor(bk, m, 32);
        if (ov < bv || (ov == bv && ok < bk)) { bv = ov; bk = ok; }
    }
    if (lane == 0) assign[row] = bk;
}

// ---- 2: normalize projections -> f16 feat (wave per row) ----
__global__ void k_normalize(const float* __restrict__ pa, const float* __restrict__ pb,
                            _Float16* __restrict__ feat) {
    int tid = threadIdx.x;
    int wave = tid >> 5, lane = tid & 31;
    int row = blockIdx.x * 8 + wave;
    const float* src = (row < BB) ? (pa + (size_t)row * DD) : (pb + (size_t)(row - BB) * DD);
    float4 v = ((const float4*)src)[lane];
    float ss = v.x * v.x + v.y * v.y + v.z * v.z + v.w * v.w;
    ss = wave_red_add(ss, 16);
    float rn = rsqrtf(fmaxf(ss, 1e-24f));
    _Float16* dst = feat + (size_t)row * DD + lane * 4;
    dst[0] = (_Float16)(v.x * rn); dst[1] = (_Float16)(v.y * rn);
    dst[2] = (_Float16)(v.z * rn); dst[3] = (_Float16)(v.w * rn);
}

// ---- 3: segment sums + counts + label histogram (global atomics) ----
__global__ void k_segsum(const float* __restrict__ pa, const float* __restrict__ pb,
                         const float* __restrict__ ea, const int* __restrict__ assign,
                         const int* __restrict__ labels,
                         float* sumA, float* sumB, float* sumS,
                         float* cntA, float* cntB, float* hist) {
    unsigned t = blockIdx.x * blockDim.x + threadIdx.x; // B*D threads
    int i = t >> 7, d = t & 127;
    int la = assign[i], lb = assign[BB + i];
    atomicAdd(&sumA[la * DD + d], pa[t]);
    atomicAdd(&sumB[lb * DD + d], pb[t]);
    atomicAdd(&sumS[la * DD + d], ea[t]);
    if (d == 0) {
        atomicAdd(&cntA[la], 1.0f);
        atomicAdd(&cntB[lb], 1.0f);
        atomicAdd(&hist[labels[i]], 1.0f);
    }
}

// ---- 4: finalize prototypes; zbuf = normalized [protoA;protoB], sup = raw mean
__global__ void k_proto(const float* sumA, const float* sumB, const float* sumS,
                        const float* cntA, const float* cntB,
                        float* zbuf, float* sup) {
    int tid = threadIdx.x;
    int g = blockIdx.x * 8 + (tid >> 5); // 192 waves
    int lane = tid & 31;
    if (g < 128) {
        int k = (g < 64) ? g : g - 64;
        const float* s = (g < 64) ? (sumA + k * DD) : (sumB + k * DD);
        float c = fmaxf((g < 64) ? cntA[k] : cntB[k], 1.0f);
        float inv = 1.0f / c;
        float4 v = ((const float4*)s)[lane];
        v.x *= inv; v.y *= inv; v.z *= inv; v.w *= inv;
        float ss = v.x * v.x + v.y * v.y + v.z * v.z + v.w * v.w;
        ss = wave_red_add(ss, 16);
        float rn = rsqrtf(fmaxf(ss, 1e-24f));
        float* dst = zbuf + (size_t)g * DD + lane * 4;
        dst[0] = v.x * rn; dst[1] = v.y * rn; dst[2] = v.z * rn; dst[3] = v.w * rn;
    } else {
        int k = g - 128;
        float inv = 1.0f / fmaxf(cntA[k], 1.0f);
        float4 v = ((const float4*)(sumS + k * DD))[lane];
        float* dst = sup + (size_t)k * DD + lane * 4;
        dst[0] = v.x * inv; dst[1] = v.y * inv; dst[2] = v.z * inv; dst[3] = v.w * inv;
    }
}

// ---- 5: l_p NT-Xent on 128x128 prototypes (one block, 128 threads) ----
__global__ void k_lp(const float* __restrict__ z, float* acc) {
    int i = threadIdx.x; // 0..127
    int pos = (i + 64) & 127;
    float m = -1e30f, s = 0.0f, vp = 0.0f;
    for (int j = 0; j < 128; ++j) {
        if (j == i) continue;
        float dot = 0.0f;
        const float* zi = z + (size_t)i * DD;
        const float* zj = z + (size_t)j * DD;
        for (int d = 0; d < DD; ++d) dot += zi[d] * zj[d];
        float v = dot * INV_T;
        if (v > m) { s = s * __expf(m - v) + 1.0f; m = v; }
        else       { s += __expf(v - m); }
        if (j == pos) vp = v;
    }
    float loss = -(vp - m - __logf(s));
    atomicAdd(&acc[0], loss);
}

// ---- 6: l_n softmax over 64 prototype distances (wave per row) ----
__global__ void k_ln(const float* __restrict__ eb, const float* __restrict__ sup,
                     const int* __restrict__ assign, float* acc) {
    __shared__ float cs[KK * DD]; // transposed
    int tid = threadIdx.x;
    async_stage_transposed(sup, tid);
    __syncthreads();
    int wave = tid >> 5, lane = tid & 31;
    int row = blockIdx.x * 8 + wave; // 0..8191
    const float* x = eb + (size_t)row * DD;
    int k0 = lane, k1 = lane + 32;
    float a0 = 0.0f, a1 = 0.0f;
    for (int d = 0; d < DD; ++d) {
        float xd = x[d];
        float f0 = xd - cs[d * KK + k0];
        float f1 = xd - cs[d * KK + k1];
        a0 += f0 * f0; a1 += f1 * f1;
    }
    float l0 = -sqrtf(fmaxf(a0, 0.0f));
    float l1 = -sqrtf(fmaxf(a1, 0.0f));
    float m = fmaxf(l0, l1);
    for (int mm = 1; mm <= 16; mm <<= 1) m = fmaxf(m, __shfl_xor(m, mm, 32));
    float s = __expf(l0 - m) + __expf(l1 - m);
    s = wave_red_add(s, 16);
    int lb = assign[BB + row];
    float tgt = (k0 == lb) ? l0 : ((k1 == lb) ? l1 : 0.0f);
    tgt = wave_red_add(tgt, 16);
    if (lane == 0) atomicAdd(&acc[1], -(tgt - m - __logf(s)));
}

// ---- 7: fused SupCon GEMM: 16-row tile per block, 8 waves split columns,
//         B tiles double-buffered in VGPRs (software pipeline) ----
__device__ __forceinline__ void load_btile(const _Float16* __restrict__ p, v16h* B) {
    #pragma unroll
    for (int c = 0; c < 4; ++c) B[c] = *(const v16h*)(p + c * 32);
}

__global__ __launch_bounds__(256)
__attribute__((amdgpu_waves_per_eu(1, 6)))
void k_supcon(const _Float16* __restrict__ feat, const int* __restrict__ labels,
              float* __restrict__ denom, float* __restrict__ pos) {
    int rbase = blockIdx.x * 16;
    int tid = threadIdx.x;
    int wave = tid >> 5, lane = tid & 31;
    int hi = lane >> 4, ln15 = lane & 15;

    // A tile (16 rows x 128 k) resident in registers: 4 chunks of k=32.
    const _Float16* arow = feat + (size_t)(rbase + ln15) * DD;
    v16h A[4];
    #pragma unroll
    for (int c = 0; c < 4; ++c) {
        int kc = c * 32;
        A[c] = *(const v16h*)(arow + kc + hi * 16);
    }
    int labi[8];
    #pragma unroll
    for (int v = 0; v < 8; ++v) labi[v] = labels[(rbase + v + 8 * hi) & (BB - 1)];

    float dAcc[8] = {0,0,0,0,0,0,0,0};
    float pAcc[8] = {0,0,0,0,0,0,0,0};

    int jt0 = wave * 128;                  // 128 column tiles of 16 per wave
    const int TSTEP = 16 * DD;             // halfs per column tile
    const _Float16* bp = feat + (size_t)(jt0 * 16 + ln15) * DD + hi * 16;

    auto epilogue = [&](int jbase, const v8f& accv) {
        int j = jbase + ln15;
        int labj = labels[j & (BB - 1)];
        #pragma unroll
        for (int v = 0; v < 8; ++v) {
            int i = rbase + v + 8 * hi;
            float sim = accv[v];
            float e = exp2f(__builtin_fmaf(sim, C_EXP, -C_EXP)); // exp((sim-1)/T)
            float notdiag = (i != j) ? 1.0f : 0.0f;
            float posmask = (labj == labi[v]) ? notdiag : 0.0f;
            dAcc[v] = __builtin_fmaf(notdiag, e, dAcc[v]);
            pAcc[v] = __builtin_fmaf(posmask, sim, pAcc[v]);
        }
    };

    v16h B0[4], B1[4];
    load_btile(bp, B0);                    // prologue: tile jt0
    // 2x-unrolled software pipeline: loads for tile t+1 issue before the
    // WMMAs + epilogue of tile t consume their buffer.
    for (int jt = jt0; jt < jt0 + 128; jt += 2) {
        load_btile(bp + TSTEP, B1);                        // tile jt+1
        __builtin_prefetch(bp + 2 * TSTEP, 0, 1);          // warm GL2 2 ahead
        {
            v8f accv = {};
            #pragma unroll
            for (int c = 0; c < 4; ++c)
                accv = __builtin_amdgcn_wmma_f32_16x16x32_f16(
                    false, A[c], false, B0[c], (short)0, accv, false, false);
            epilogue(jt * 16, accv);
        }
        load_btile(bp + 2 * TSTEP, B0);                    // tile jt+2 (last iter:
        __builtin_prefetch(bp + 3 * TSTEP, 0, 1);          //  harmless re-read)
        {
            v8f accv = {};
            #pragma unroll
            for (int c = 0; c < 4; ++c)
                accv = __builtin_amdgcn_wmma_f32_16x16x32_f16(
                    false, A[c], false, B1[c], (short)0, accv, false, false);
            epilogue(jt * 16 + 16, accv);
        }
        bp += 2 * TSTEP;
    }

    #pragma unroll
    for (int v = 0; v < 8; ++v) {
        float dv = dAcc[v], pv = pAcc[v];
        for (int m = 1; m <= 8; m <<= 1) {
            dv += __shfl_xor(dv, m, 32);
            pv += __shfl_xor(pv, m, 32);
        }
        if (ln15 == 0) {
            int row = rbase + v + 8 * hi;
            atomicAdd(&denom[row], dv);
            atomicAdd(&pos[row], pv);
        }
    }
}

// ---- 8: final reduction into scalar loss ----
__global__ void k_final(const float* __restrict__ denom, const float* __restrict__ pos,
                        const float* __restrict__ hist, const int* __restrict__ labels,
                        const float* __restrict__ acc, float* __restrict__ out) {
    __shared__ float red[256];
    int tid = threadIdx.x;
    float lm = 0.0f;
    for (int i = tid; i < NN; i += 256) {
        float c = 2.0f * hist[labels[i & (BB - 1)]] - 1.0f;
        lm += (pos[i] - c) * INV_T / c - __logf(denom[i]);
    }
    red[tid] = lm;
    __syncthreads();
    for (int s = 128; s > 0; s >>= 1) {
        if (tid < s) red[tid] += red[tid + s];
        __syncthreads();
    }
    if (tid == 0) {
        float l_main = -red[0] / (float)NN; // temp/base_temp == 1
        float l_p = acc[0] / 128.0f;
        float l_n = acc[1] / (float)BB;
        out[0] = l_main + 0.5f * l_p + 0.5f * l_n;
    }
}

extern "C" void kernel_launch(void* const* d_in, const int* in_sizes, int n_in,
                              void* d_out, int out_size, void* d_ws, size_t ws_size,
                              hipStream_t stream) {
    const float* ea = (const float*)d_in[0];
    const float* eb = (const float*)d_in[1];
    const float* pa = (const float*)d_in[2];
    const float* pb = (const float*)d_in[3];
    const float* ct = (const float*)d_in[4];
    const int*   lb = (const int*)  d_in[5];
    float* out = (float*)d_out;

    char* ws = (char*)d_ws;
    _Float16* feat = (_Float16*)(ws + OFF_FEAT);
    int*   assign  = (int*)  (ws + OFF_ASSIGN);
    float* sumA    = (float*)(ws + OFF_SUMA);
    float* sumB    = (float*)(ws + OFF_SUMB);
    float* sumS    = (float*)(ws + OFF_SUMS);
    float* cntA    = (float*)(ws + OFF_CNTA);
    float* cntB    = (float*)(ws + OFF_CNTB);
    float* hist    = (float*)(ws + OFF_HIST);
    float* denom   = (float*)(ws + OFF_DENOM);
    float* posb    = (float*)(ws + OFF_POS);
    float* acc     = (float*)(ws + OFF_ACC);
    float* zbuf    = (float*)(ws + OFF_ZBUF);
    float* sup     = (float*)(ws + OFF_SUP);

    k_zero<<<(ZERO_FLOATS + 255) / 256, 256, 0, stream>>>((float*)(ws + OFF_SUMA), ZERO_FLOATS);
    k_assign<<<NN / 8, 256, 0, stream>>>(ea, eb, ct, assign);
    k_normalize<<<NN / 8, 256, 0, stream>>>(pa, pb, feat);
    k_segsum<<<(BB * DD) / 256, 256, 0, stream>>>(pa, pb, ea, assign, lb,
                                                  sumA, sumB, sumS, cntA, cntB, hist);
    k_proto<<<24, 256, 0, stream>>>(sumA, sumB, sumS, cntA, cntB, zbuf, sup);
    k_lp<<<1, 128, 0, stream>>>(zbuf, acc);
    k_ln<<<BB / 8, 256, 0, stream>>>(eb, sup, assign, acc);
    k_supcon<<<NN / 16, 256, 0, stream>>>(feat, lb, denom, posb);
    k_final<<<1, 256, 0, stream>>>(denom, posb, hist, lb, acc, out);
}
